// MultiheadAttention_46789373723020
// MI455X (gfx1250) — matrix operand order repaired and checked by reading, compiled, not run on hardware
//
#include <hip/hip_runtime.h>
#include <stdint.h>

// MI455X / gfx1250 multi-head attention, bf16 WMMA path.
// B=2, T=S=4096, D=512, H=8, Dh=64.

typedef __attribute__((ext_vector_type(16))) __bf16 v16bf;
typedef __attribute__((ext_vector_type(8)))  float  v8f;

#define B_   2
#define H_   8
#define T_   4096
#define S_   4096
#define D_   512
#define DH_  64
#define NROWS (B_ * T_)
#define SCALE 0.125f   // 1/sqrt(64)

struct Frag {
    union {
        v16bf v;
        __bf16 h[16];
        uint4 q[2];
    };
};

// Native RNE conversion; lowers to the hardware f32->bf16 path.
__device__ __forceinline__ __bf16 f2bf(float f) { return (__bf16)f; }

__device__ __forceinline__ uint4 ld16(const void* p) { return *(const uint4*)p; }

__device__ __forceinline__ v8f wmma_bf16(const Frag& a, const Frag& b, v8f c) {
    return __builtin_amdgcn_wmma_f32_16x16x32_bf16(
        /*neg_a=*/false, a.v, /*neg_b=*/false, b.v,
        /*c_mod=*/(short)0, c, /*reuse_a=*/false, /*reuse_b=*/false);
}

// ---------------------------------------------------------------------------
// Kernel 1: fused projections.  Out[n,d] = X[n,:] . W[d,:]  (+bias, *scale for Q)
// z = blockIdx.z selects {Q,K,V}.  One wave computes a 16(n) x 64(d) tile.
// Q,K stored bf16 [B,H,T,Dh]; V stored bf16 transposed [B,H,Dh,S].
// ---------------------------------------------------------------------------
__global__ void proj_kernel(const float* __restrict__ Xq, const float* __restrict__ Xk,
                            const float* __restrict__ Xv,
                            const float* __restrict__ Wq, const float* __restrict__ Wk,
                            const float* __restrict__ Wv, const float* __restrict__ bq,
                            __bf16* __restrict__ Qo, __bf16* __restrict__ Ko,
                            __bf16* __restrict__ Vt)
{
    const int lane = threadIdx.x & 31;
    const int lh   = lane & 15;
    const int half = lane >> 4;
    const int z    = blockIdx.z;
    const int n0   = blockIdx.x * 16;
    const int dsup = blockIdx.y * 64;

    const float* X = (z == 0) ? Xq : (z == 1) ? Xk : Xv;
    const float* W = (z == 0) ? Wq : (z == 1) ? Wk : Wv;

    v8f zero = {0.f,0.f,0.f,0.f,0.f,0.f,0.f,0.f};
    v8f acc[4] = {zero, zero, zero, zero};

    const float* Arow = X + (size_t)(n0 + lh) * D_;

    for (int k0 = 0; k0 < D_; k0 += 32) {
        // A fragment: row n = n0+lh, K chunks [kA..kA+7] and [kA+16..kA+23]
        Frag a;
        {
            const float* p = Arow + k0 + half * 8;
            float4 f0 = *(const float4*)(p);
            float4 f1 = *(const float4*)(p + 4);
            float4 f2 = *(const float4*)(p + 16);
            float4 f3 = *(const float4*)(p + 20);
            a.h[0]=f2bf(f0.x);  a.h[1]=f2bf(f0.y);  a.h[2]=f2bf(f0.z);  a.h[3]=f2bf(f0.w);
            a.h[4]=f2bf(f1.x);  a.h[5]=f2bf(f1.y);  a.h[6]=f2bf(f1.z);  a.h[7]=f2bf(f1.w);
            a.h[8]=f2bf(f2.x);  a.h[9]=f2bf(f2.y);  a.h[10]=f2bf(f2.z); a.h[11]=f2bf(f2.w);
            a.h[12]=f2bf(f3.x); a.h[13]=f2bf(f3.y); a.h[14]=f2bf(f3.z); a.h[15]=f2bf(f3.w);
        }
#pragma unroll
        for (int dt = 0; dt < 4; ++dt) {
            // B fragment: B[k][d] = W[d][k]; lane holds col d, K chunks [kB..kB+7] and [kB+16..kB+23]
            Frag b;
            const float* p = W + (size_t)(dsup + dt * 16 + lh) * D_ + k0 + half * 8;
            float4 g0 = *(const float4*)(p);
            float4 g1 = *(const float4*)(p + 4);
            float4 g2 = *(const float4*)(p + 16);
            float4 g3 = *(const float4*)(p + 20);
            b.h[0]=f2bf(g0.x);  b.h[1]=f2bf(g0.y);  b.h[2]=f2bf(g0.z);  b.h[3]=f2bf(g0.w);
            b.h[4]=f2bf(g1.x);  b.h[5]=f2bf(g1.y);  b.h[6]=f2bf(g1.z);  b.h[7]=f2bf(g1.w);
            b.h[8]=f2bf(g2.x);  b.h[9]=f2bf(g2.y);  b.h[10]=f2bf(g2.z); b.h[11]=f2bf(g2.w);
            b.h[12]=f2bf(g3.x); b.h[13]=f2bf(g3.y); b.h[14]=f2bf(g3.z); b.h[15]=f2bf(g3.w);
            acc[dt] = wmma_bf16(a, b, acc[dt]);
        }
    }

    const int bI    = n0 >> 12;          // batch (T_ == 4096)
    const int tbase = n0 & (T_ - 1);

#pragma unroll
    for (int dt = 0; dt < 4; ++dt) {
        const int d  = dsup + dt * 16 + lh;
        const int hh = d >> 6;
        const int dh = d & 63;
        if (z == 2) {
            // V transposed: Vt[b][h][dh][s]; lane's 8 rows are 8 consecutive s.
            union { __bf16 hs[8]; uint4 q; } pk;
#pragma unroll
            for (int v = 0; v < 8; ++v) pk.hs[v] = f2bf(acc[dt][v]);
            size_t base = ((size_t)(bI * H_ + hh) * DH_ + dh) * S_ + (tbase + half * 8);
            *(uint4*)(Vt + base) = pk.q;
        } else {
            const float bias = (z == 0) ? bq[d] : 0.f;
            __bf16* O = (z == 0) ? Qo : Ko;
#pragma unroll
            for (int v = 0; v < 8; ++v) {
                const int t = tbase + v + half * 8;
                float y = acc[dt][v] + bias;
                if (z == 0) y *= SCALE;            // fold softmax scale into Q
                O[((size_t)(bI * H_ + hh) * T_ + t) * DH_ + dh] = f2bf(y);
            }
        }
    }
}

// ---------------------------------------------------------------------------
// Kernel 2: streaming online-softmax statistics (row max m, denominator l).
// One wave per (b,h, 16-row t-tile); loops all 256 s-tiles with 2 WMMAs each.
// ---------------------------------------------------------------------------
__global__ void stats_kernel(const __bf16* __restrict__ Qb,
                             const __bf16* __restrict__ Kb,
                             float* __restrict__ Mr, float* __restrict__ Lr)
{
    const int lane = threadIdx.x & 31;
    const int lh   = lane & 15;
    const int half = lane >> 4;
    const int bh   = blockIdx.y;
    const int t0   = blockIdx.x * 16;

    const __bf16* Q = Qb + ((size_t)bh * T_ + t0) * DH_;
    const __bf16* K = Kb + (size_t)bh * S_ * DH_;

    Frag aq[2];
#pragma unroll
    for (int f = 0; f < 2; ++f) {
        const int kA = f * 32 + half * 8;
        const __bf16* p = Q + lh * DH_ + kA;
        aq[f].q[0] = ld16(p);
        aq[f].q[1] = ld16(p + 16);
    }

    float m[8], l[8];
#pragma unroll
    for (int v = 0; v < 8; ++v) { m[v] = -3.4e38f; l[v] = 0.f; }

    v8f zero = {0.f,0.f,0.f,0.f,0.f,0.f,0.f,0.f};
    for (int s0 = 0; s0 < S_; s0 += 16) {
        Frag b0, b1;
        const __bf16* pr = K + (size_t)(s0 + lh) * DH_;
        b0.q[0] = ld16(pr + half * 8);
        b0.q[1] = ld16(pr + half * 8 + 16);
        b1.q[0] = ld16(pr + 32 + half * 8);
        b1.q[1] = ld16(pr + 32 + half * 8 + 16);
        v8f c = wmma_bf16(aq[0], b0, zero);
        c = wmma_bf16(aq[1], b1, c);
#pragma unroll
        for (int v = 0; v < 8; ++v) {
            const float x = c[v];
            float r = x;
            r = fmaxf(r, __shfl_xor(r, 1));
            r = fmaxf(r, __shfl_xor(r, 2));
            r = fmaxf(r, __shfl_xor(r, 4));
            r = fmaxf(r, __shfl_xor(r, 8));
            const float mn = fmaxf(m[v], r);
            float e = __expf(x - mn);
            e += __shfl_xor(e, 1);
            e += __shfl_xor(e, 2);
            e += __shfl_xor(e, 4);
            e += __shfl_xor(e, 8);
            l[v] = l[v] * __expf(m[v] - mn) + e;
            m[v] = mn;
        }
    }
    if (lh == 0) {
#pragma unroll
        for (int v = 0; v < 8; ++v) {
            const int row = t0 + v + half * 8;
            Mr[(size_t)bh * T_ + row] = m[v];
            Lr[(size_t)bh * T_ + row] = l[v];
        }
    }
}

// ---------------------------------------------------------------------------
// Kernel 3: recompute scores, normalize, write probs (fp32, once, NT), and
// accumulate P@V via WMMA (prob tile re-laid out D->A through LDS).
// ---------------------------------------------------------------------------
__global__ void attn_kernel(const __bf16* __restrict__ Qb,
                            const __bf16* __restrict__ Kb,
                            const __bf16* __restrict__ Vt,
                            const float* __restrict__ Mr, const float* __restrict__ Lr,
                            float* __restrict__ outO, float* __restrict__ outW)
{
    __shared__ __bf16 sP[16 * 32];

    const int lane = threadIdx.x & 31;
    const int lh   = lane & 15;
    const int half = lane >> 4;
    const int bh   = blockIdx.y;
    const int t0   = blockIdx.x * 16;

    const __bf16* Q = Qb + ((size_t)bh * T_ + t0) * DH_;
    const __bf16* K = Kb + (size_t)bh * S_ * DH_;
    const __bf16* V = Vt + (size_t)bh * DH_ * S_;

    Frag aq[2];
#pragma unroll
    for (int f = 0; f < 2; ++f) {
        const int kA = f * 32 + half * 8;
        const __bf16* p = Q + lh * DH_ + kA;
        aq[f].q[0] = ld16(p);
        aq[f].q[1] = ld16(p + 16);
    }

    float mr[8], li[8];
#pragma unroll
    for (int v = 0; v < 8; ++v) {
        const int row = t0 + v + half * 8;
        mr[v] = Mr[(size_t)bh * T_ + row];
        li[v] = 1.f / Lr[(size_t)bh * T_ + row];
    }

    v8f zero = {0.f,0.f,0.f,0.f,0.f,0.f,0.f,0.f};
    v8f acc[4] = {zero, zero, zero, zero};
    float* Wrow = outW + (size_t)bh * T_ * S_;

    for (int s0 = 0; s0 < S_; s0 += 32) {
#pragma unroll
        for (int sub = 0; sub < 2; ++sub) {
            const int s1 = s0 + sub * 16;
            Frag b0, b1;
            const __bf16* pr = K + (size_t)(s1 + lh) * DH_;
            b0.q[0] = ld16(pr + half * 8);
            b0.q[1] = ld16(pr + half * 8 + 16);
            b1.q[0] = ld16(pr + 32 + half * 8);
            b1.q[1] = ld16(pr + 32 + half * 8 + 16);
            v8f c = wmma_bf16(aq[0], b0, zero);
            c = wmma_bf16(aq[1], b1, c);
#pragma unroll
            for (int v = 0; v < 8; ++v) {
                const float p = __expf(c[v] - mr[v]) * li[v];
                // streaming 1 GB output: non-temporal, don't thrash L2
                __builtin_nontemporal_store(
                    p, &Wrow[(size_t)(t0 + v + half * 8) * S_ + s1 + lh]);
                sP[(v + half * 8) * 32 + sub * 16 + lh] = f2bf(p);   // stage for PV
            }
        }
        __syncthreads();
        // Prob tile as A fragment (16 t x 32 s)
        Frag pa;
        {
            const int kA = half * 8;
            pa.q[0] = ld16(&sP[lh * 32 + kA]);
            pa.q[1] = ld16(&sP[lh * 32 + kA + 16]);
        }
#pragma unroll
        for (int dt = 0; dt < 4; ++dt) {
            Frag vb;   // B[k=s][n=dh] = Vt[dh][s0+k] : k chunks [8h..8h+7] and [16+8h..16+8h+7]
            const __bf16* p = V + (size_t)(dt * 16 + lh) * S_ + s0 + half * 8;
            vb.q[0] = ld16(p);
            vb.q[1] = ld16(p + 16);
            acc[dt] = wmma_bf16(pa, vb, acc[dt]);
        }
        __syncthreads();
    }

    // attn_output layout [B, T, H, Dh]
    const int bI = bh >> 3;
    const int hh = bh & 7;
#pragma unroll
    for (int dt = 0; dt < 4; ++dt) {
        const int dh = dt * 16 + lh;
#pragma unroll
        for (int v = 0; v < 8; ++v) {
            const int t = t0 + v + half * 8;
            __builtin_nontemporal_store(
                acc[dt][v], &outO[(((size_t)bI * T_ + t) * H_ + hh) * DH_ + dh]);
        }
    }
}

// ---------------------------------------------------------------------------
extern "C" void kernel_launch(void* const* d_in, const int* in_sizes, int n_in,
                              void* d_out, int out_size, void* d_ws, size_t ws_size,
                              hipStream_t stream)
{
    (void)in_sizes; (void)n_in; (void)out_size; (void)ws_size;

    const float* q  = (const float*)d_in[0];
    const float* k  = (const float*)d_in[1];
    const float* v  = (const float*)d_in[2];
    const float* Wq = (const float*)d_in[3];
    const float* bq = (const float*)d_in[4];
    const float* Wk = (const float*)d_in[5];
    const float* Wv = (const float*)d_in[6];
    float* out = (float*)d_out;

    const size_t QKV = (size_t)B_ * H_ * T_ * DH_;   // 4,194,304 elements

    __bf16* Qbf = (__bf16*)d_ws;                     // bf16 [B,H,T,Dh]   8 MB
    __bf16* Kbf = Qbf + QKV;                         // bf16 [B,H,S,Dh]   8 MB
    __bf16* Vtr = Kbf + QKV;                         // bf16 [B,H,Dh,S]   8 MB
    float* Mr = (float*)(Vtr + QKV);                 // [B*H*T] row max
    float* Lr = Mr + (size_t)B_ * H_ * T_;           // [B*H*T] row denom

    float* outO = out;        // attn_output: B*T*H*Dh floats
    float* outW = out + QKV;  // attn_weights: B*H*T*S floats

    proj_kernel<<<dim3(NROWS / 16, D_ / 64, 3), 32, 0, stream>>>(
        q, k, v, Wq, Wk, Wv, bq, Qbf, Kbf, Vtr);
    stats_kernel<<<dim3(T_ / 16, B_ * H_), 32, 0, stream>>>(Qbf, Kbf, Mr, Lr);
    attn_kernel<<<dim3(T_ / 16, B_ * H_), 32, 0, stream>>>(Qbf, Kbf, Vtr, Mr, Lr, outO, outW);
}
